// EEM_71571335020895
// MI455X (gfx1250) — matrix operand order repaired and checked by reading, not compiled, not run on hardware
//
#include <hip/hip_runtime.h>

#define B_N 8
#define COILS 16
#define HGT 384
#define WID 384
#define HW (HGT*WID)
#define C2 192
#define EPSF 1e-12f

#define TY 32
#define TX 32
#define HALO_W (TX+2)
#define HALO_H (TY+2)
#define HALO_SZ (HALO_W*HALO_H)          // 1156
#define NPIX (TY*TX)                     // 1024
#define HALO_FLOATS (3*HALO_SZ)          // 3468
#define HALO_BYTES  (HALO_FLOATS*4)      // 13872 (16B aligned)
#define BAS_HALFS (32*NPIX)              // 32768 halfs = 64KB
#define SMEM_BYTES (HALO_BYTES + BAS_HALFS*2)

typedef _Float16 half_t;
typedef __attribute__((ext_vector_type(8)))  _Float16 v8h;
typedef __attribute__((ext_vector_type(16))) _Float16 v16h;
typedef __attribute__((ext_vector_type(8)))  float    v8f;

// ---------------- kernel 1: coil combine: red = sum_c cmul(x_c, conj(sens_c)) ----
__global__ void k_coil_combine(const float* __restrict__ x,
                               const float* __restrict__ sens,
                               float* __restrict__ ximg) {
  int pix = blockIdx.x * blockDim.x + threadIdx.x;
  int b = blockIdx.y;
  if (pix >= HW) return;
  const float* xb = x    + (size_t)b * COILS * HW * 2 + (size_t)pix * 2;
  const float* sb = sens + (size_t)b * COILS * HW * 2 + (size_t)pix * 2;
  float r0 = 0.f, r1 = 0.f;
  for (int c = 0; c < COILS; ++c) {
    const float2 a = *(const float2*)(xb + (size_t)c * HW * 2);
    const float2 s = *(const float2*)(sb + (size_t)c * HW * 2);
    if (c + 1 < COILS) {
      __builtin_prefetch(xb + (size_t)(c + 1) * HW * 2, 0, 1);
      __builtin_prefetch(sb + (size_t)(c + 1) * HW * 2, 0, 1);
    }
    r0 += a.x * s.x + a.y * s.y;   // re of a * conj(s)
    r1 += a.y * s.x - a.x * s.y;   // im of a * conj(s)
  }
  ximg[(size_t)(b * 2 + 0) * HW + pix] = r0;
  ximg[(size_t)(b * 2 + 1) * HW + pix] = r1;
}

// ---------------- kernel 2: 27x27 (padded 32x32) Gram of shifted basis fields ----
// basis row = i*9 + t, i in {re, im, e}, tap t -> (dy,dx) in {0,1,2}^2 relative
// to the halo origin; rows 27..31 are zero-filled so fragment loads never branch.
__global__ void __launch_bounds__(256)
k_gram27(const float* __restrict__ ximg, const float* __restrict__ e,
         float* __restrict__ G) {
  __shared__ __align__(16) char smem[SMEM_BYTES];
  float*  halo = (float*)smem;
  half_t* bas  = (half_t*)(smem + HALO_BYTES);
  float*  sred = (float*)smem;           // aliases halo+bas after they are dead

  int b = blockIdx.y;
  int tyi = blockIdx.x / (WID / TX);
  int txi = blockIdx.x % (WID / TX);
  int y0 = tyi * TY, x0 = txi * TX;
  const float* re_p = ximg + (size_t)(b * 2 + 0) * HW;
  const float* im_p = ximg + (size_t)(b * 2 + 1) * HW;
  const float* e_p  = e    + (size_t)b * HW;

  for (int fi = threadIdx.x; fi < HALO_FLOATS; fi += 256) {
    int f = fi / HALO_SZ;
    int rem = fi - f * HALO_SZ;
    int hy = rem / HALO_W, hx = rem - hy * HALO_W;
    int gy = y0 + hy - 1, gx = x0 + hx - 1;
    float v = 0.f;
    if (gy >= 0 && gy < HGT && gx >= 0 && gx < WID) {
      const float* src = (f == 0) ? re_p : (f == 1) ? im_p : e_p;
      v = src[gy * WID + gx];
    }
    halo[fi] = v;
  }
  __syncthreads();

  // stage f16 basis matrix bas[row][pixel], branch-free
  #pragma unroll 4
  for (int idx = threadIdx.x; idx < 27 * NPIX; idx += 256) {
    int row = idx >> 10;
    int pp  = idx & (NPIX - 1);
    int f   = row / 9;
    int tap = row - f * 9;
    int dy = tap / 3, dx = tap - (tap / 3) * 3;
    int ly = pp >> 5, lx = pp & 31;
    bas[idx] = (half_t)halo[f * HALO_SZ + (ly + dy) * HALO_W + (lx + dx)];
  }
  for (int idx = 27 * NPIX + threadIdx.x; idx < 32 * NPIX; idx += 256)
    bas[idx] = (half_t)0.f;
  __syncthreads();

  int lane = threadIdx.x & 31;
  int wave = threadIdx.x >> 5;
  int grp  = lane >> 4;                  // lane group 0/1
  int l16  = lane & 15;
  int base = wave * 128;                 // 128 pixels per wave

  const v8h* basv = (const v8h*)bas;     // 16B granules, 128 per basis row
  v8f acc00 = {}, acc01 = {}, acc10 = {}, acc11 = {};
  #pragma unroll
  for (int cc = 0; cc < 4; ++cc) {       // 4 chunks of K=32 pixels
    int kbase = base + cc * 32;          // multiple of 32
    int gA = (kbase >> 3) + grp;         // A: K = kbase + grp*8 (+16 for hi half)
    int gB = (kbase >> 3) + grp;         // B: K = kbase + grp*8 (+16 for hi half)
    v8h a0lo = basv[l16 * 128 + gA];
    v8h a0hi = basv[l16 * 128 + gA + 2];
    v8h a1lo = basv[(16 + l16) * 128 + gA];
    v8h a1hi = basv[(16 + l16) * 128 + gA + 2];
    v8h b0lo = basv[l16 * 128 + gB];
    v8h b0hi = basv[l16 * 128 + gB + 2];
    v8h b1lo = basv[(16 + l16) * 128 + gB];
    v8h b1hi = basv[(16 + l16) * 128 + gB + 2];
    v16h A0 = __builtin_shufflevector(a0lo, a0hi, 0,1,2,3,4,5,6,7,8,9,10,11,12,13,14,15);
    v16h A1 = __builtin_shufflevector(a1lo, a1hi, 0,1,2,3,4,5,6,7,8,9,10,11,12,13,14,15);
    v16h B0 = __builtin_shufflevector(b0lo, b0hi, 0,1,2,3,4,5,6,7,8,9,10,11,12,13,14,15);
    v16h B1 = __builtin_shufflevector(b1lo, b1hi, 0,1,2,3,4,5,6,7,8,9,10,11,12,13,14,15);
    acc00 = __builtin_amdgcn_wmma_f32_16x16x32_f16(false, A0, false, B0, (short)0, acc00, false, false);
    acc01 = __builtin_amdgcn_wmma_f32_16x16x32_f16(false, A0, false, B1, (short)0, acc01, false, false);
    acc10 = __builtin_amdgcn_wmma_f32_16x16x32_f16(false, A1, false, B0, (short)0, acc10, false, false);
    acc11 = __builtin_amdgcn_wmma_f32_16x16x32_f16(false, A1, false, B1, (short)0, acc11, false, false);
  }
  __syncthreads();                        // bas dead; safe to alias with sred

  // C/D layout: VGPR r -> M = r + 8*grp, N = lane%16
  float* wr = sred + wave * 1024;
  #pragma unroll
  for (int r = 0; r < 8; ++r) {
    int m = r + grp * 8;
    wr[m * 32 + l16]             = acc00[r];
    wr[m * 32 + 16 + l16]        = acc01[r];
    wr[(16 + m) * 32 + l16]      = acc10[r];
    wr[(16 + m) * 32 + 16 + l16] = acc11[r];
  }
  __syncthreads();
  for (int eidx = threadIdx.x; eidx < 1024; eidx += 256) {
    float s = 0.f;
    #pragma unroll
    for (int w = 0; w < 8; ++w) s += sred[w * 1024 + eidx];
    atomicAdd(&G[b * 1024 + eidx], s);
  }
}

// ---------------- kernel 3: attention statistics -> effective 2x18 conv weights ----
__global__ void __launch_bounds__(192)
k_attn(const float* __restrict__ G, const float* __restrict__ w_im1,
       const float* __restrict__ w_im2, const float* __restrict__ w_e,
       const float* __restrict__ w_proj, const float* __restrict__ a1,
       float* __restrict__ Weff) {
  __shared__ float Gs[32 * 32];
  __shared__ float nk_s[192];
  __shared__ float wacc[36];
  int b = blockIdx.x;
  int tid = threadIdx.x;                 // tid = h*24 + c
  for (int i = tid; i < 1024; i += 192) Gs[i] = G[b * 1024 + i];
  if (tid < 36) wacc[tid] = 0.f;
  __syncthreads();

  int h = tid / 24;
  int chq = tid;

  float aq[18];                          // q coefficients over {re,im} x 9 taps
  for (int t = 0; t < 9; ++t) {
    float w2 = w_im2[chq * 9 + t];
    aq[t]     = w2 * w_im1[chq * 2 + 0];
    aq[9 + t] = w2 * w_im1[chq * 2 + 1];
  }
  float nq2 = 0.f;                       // ||q||^2 = aq^T G_ff aq
  for (int r = 0; r < 18; ++r) {
    float u = 0.f;
    for (int r2 = 0; r2 < 18; ++r2) u += Gs[r * 32 + r2] * aq[r2];
    nq2 += aq[r] * u;
  }
  float nq = fmaxf(sqrtf(fmaxf(nq2, 0.f)), EPSF);
  {                                      // ||k_d||^2 for d = c: ge^T G_ee ge
    float nk2 = 0.f;
    for (int s = 0; s < 9; ++s) {
      float u = 0.f;
      for (int s2 = 0; s2 < 9; ++s2) u += Gs[(18 + s) * 32 + 18 + s2] * w_e[chq * 9 + s2];
      nk2 += w_e[chq * 9 + s] * u;
    }
    nk_s[chq] = fmaxf(sqrtf(fmaxf(nk2, 0.f)), EPSF);
  }
  __syncthreads();

  float tq[9];                           // tq[s] = aq^T G[:,18+s]
  for (int s = 0; s < 9; ++s) {
    float u = 0.f;
    for (int r = 0; r < 18; ++r) u += aq[r] * Gs[r * 32 + 18 + s];
    tq[s] = u;
  }
  float scale = a1[h];
  float logit[24];
  float mx = -3.4e38f;
  for (int d = 0; d < 24; ++d) {
    int chk = h * 24 + d;
    float S = 0.f;
    for (int s = 0; s < 9; ++s) S += tq[s] * w_e[chk * 9 + s];
    float l = scale * S / (nq * nk_s[chk]);
    logit[d] = l;
    mx = fmaxf(mx, l);
  }
  float sum = 0.f;
  for (int d = 0; d < 24; ++d) { logit[d] = __expf(logit[d] - mx); sum += logit[d]; }
  float inv = 1.f / sum;

  float tmp[18];                         // attn row folded into v basis coeffs
  for (int r = 0; r < 18; ++r) tmp[r] = 0.f;
  for (int d = 0; d < 24; ++d) {
    float a = logit[d] * inv;
    int chv = C2 + h * 24 + d;
    float w10 = w_im1[chv * 2 + 0], w11 = w_im1[chv * 2 + 1];
    for (int t = 0; t < 9; ++t) {
      float w2 = w_im2[chv * 9 + t];
      tmp[t]     += a * w2 * w10;
      tmp[9 + t] += a * w2 * w11;
    }
  }
  float p0 = w_proj[0 * C2 + chq];
  float p1 = w_proj[1 * C2 + chq];
  for (int r = 0; r < 18; ++r) {
    atomicAdd(&wacc[r],      p0 * tmp[r]);
    atomicAdd(&wacc[18 + r], p1 * tmp[r]);
  }
  __syncthreads();
  if (tid < 36) Weff[b * 36 + tid] = wacc[tid];
}

// ---------------- kernel 4: effective 3x3 conv + residual + coil expand ----------
__global__ void k_final(const float* __restrict__ ximg, const float* __restrict__ Weff,
                        const float* __restrict__ sens, float* __restrict__ out) {
  int pix = blockIdx.x * blockDim.x + threadIdx.x;
  int b = blockIdx.y;
  if (pix >= HW) return;
  int y = pix / WID, x = pix - y * WID;
  const float* re = ximg + (size_t)(b * 2 + 0) * HW;
  const float* im = ximg + (size_t)(b * 2 + 1) * HW;
  const float* We = Weff + b * 36;
  float o0 = re[pix], o1 = im[pix];      // residual
  for (int t = 0; t < 9; ++t) {
    int gy = y + t / 3 - 1, gx = x + t % 3 - 1;
    float fre = 0.f, fim = 0.f;
    if (gy >= 0 && gy < HGT && gx >= 0 && gx < WID) {
      fre = re[gy * WID + gx];
      fim = im[gy * WID + gx];
    }
    o0 += We[t] * fre + We[9 + t] * fim;
    o1 += We[18 + t] * fre + We[27 + t] * fim;
  }
  const float* sb = sens + (size_t)b * COILS * HW * 2 + (size_t)pix * 2;
  float* ob = out + (size_t)b * COILS * HW * 2 + (size_t)pix * 2;
  for (int c = 0; c < COILS; ++c) {
    float2 s = *(const float2*)(sb + (size_t)c * HW * 2);
    float2 r;
    r.x = o0 * s.x - o1 * s.y;           // cmul(o, s)
    r.y = o0 * s.y + o1 * s.x;
    *(float2*)(ob + (size_t)c * HW * 2) = r;
  }
}

extern "C" void kernel_launch(void* const* d_in, const int* in_sizes, int n_in,
                              void* d_out, int out_size, void* d_ws, size_t ws_size,
                              hipStream_t stream) {
  const float* x      = (const float*)d_in[0];
  const float* e      = (const float*)d_in[1];
  const float* sens   = (const float*)d_in[2];
  const float* w_im1  = (const float*)d_in[3];
  const float* w_im2  = (const float*)d_in[4];
  const float* w_e    = (const float*)d_in[5];
  const float* w_proj = (const float*)d_in[6];
  const float* a1     = (const float*)d_in[7];
  float* out = (float*)d_out;

  float* ximg = (float*)d_ws;                          // B*2*HW floats (9.4 MB)
  float* G    = ximg + (size_t)B_N * 2 * HW;           // B*1024 floats
  float* Weff = G + (size_t)B_N * 1024;                // B*36 floats

  hipMemsetAsync(G, 0, (size_t)B_N * 1024 * sizeof(float), stream);

  dim3 blk(256);
  dim3 g1((HW + 255) / 256, B_N);
  k_coil_combine<<<g1, blk, 0, stream>>>(x, sens, ximg);

  dim3 g2((HGT / TY) * (WID / TX), B_N);
  k_gram27<<<g2, dim3(256), 0, stream>>>(ximg, e, G);

  k_attn<<<dim3(B_N), dim3(192), 0, stream>>>(G, w_im1, w_im2, w_e, w_proj, a1, Weff);

  k_final<<<g1, blk, 0, stream>>>(ximg, Weff, sens, out);
}
